// CausalSelfAttention_47416438948538
// MI455X (gfx1250) — compile-verified, matrix operand order checked by reading, not run on hardware
//
#include <hip/hip_runtime.h>
#include <cstdint>

typedef unsigned short bf16_t;
typedef unsigned int u32;
typedef __attribute__((ext_vector_type(4)))  u32    u32x4;
typedef __attribute__((ext_vector_type(8)))  u32    u32x8;
typedef __attribute__((ext_vector_type(16))) __bf16 v16bf;
typedef __attribute__((ext_vector_type(8)))  float  v8f;

// ---------- helpers ----------
__device__ __forceinline__ bf16_t f2b(float f) {
  u32 u = __builtin_bit_cast(u32, f);
  u += 0x7FFFu + ((u >> 16) & 1u);   // round-to-nearest-even
  return (bf16_t)(u >> 16);
}

struct FragBits { u32x4 lo, hi; };

__device__ __forceinline__ v16bf frag_cast(u32x4 lo, u32x4 hi) {
  FragBits p; p.lo = lo; p.hi = hi;
  return __builtin_bit_cast(v16bf, p);
}

__device__ __forceinline__ v16bf load_frag(const bf16_t* rowp, int kb) {
  // rowp points at element k=0 of this lane's row; kb = (lane>=16)?8:0
  return frag_cast(*(const u32x4*)(rowp + kb), *(const u32x4*)(rowp + kb + 16));
}

__device__ __forceinline__ v8f wmma_bf16(v16bf a, v16bf b, v8f c) {
  return __builtin_amdgcn_wmma_f32_16x16x32_bf16(false, a, false, b, (short)0, c,
                                                 false, false);
}

// ---------- Tensor Data Mover: 2-D bf16 tile (rows x rowlen) -> LDS ----------
// D# packing per CDNA5 ISA:
//   group0: count=1 | lds_addr | global_addr[56:0] | type=2
//   group1: data_size=2B, tensor_dim0=rowlen, tensor_dim1=rows,
//           tile_dim0=rowlen, tile_dim1=rows, tensor_dim0_stride=stride
__device__ __forceinline__ void tdm_load_2d(u32 lds_off, const bf16_t* gptr,
                                            u32 rows, u32 rowlen, u32 stride_elems) {
  unsigned long long ga = (unsigned long long)(uintptr_t)gptr;
  u32x4 g0;
  g0[0] = 1u;                                                 // count = 1
  g0[1] = lds_off;                                            // lds_addr (bytes)
  g0[2] = (u32)ga;                                            // global_addr lo
  g0[3] = ((u32)(ga >> 32) & 0x01FFFFFFu) | 0x80000000u;      // ga hi | type=2
  u32x8 g1;
  g1[0] = 0x00010000u;                                        // data_size = 2 bytes
  g1[1] = (rowlen & 0xFFFFu) << 16;                           // tensor_dim0[15:0]
  g1[2] = (rows & 0xFFFFu) << 16;                             // td0 hi=0 | tensor_dim1[15:0]
  g1[3] = (rowlen & 0xFFFFu) << 16;                           // td1 hi=0 | tile_dim0
  g1[4] = rows & 0xFFFFu;                                     // tile_dim1 | tile_dim2=0
  g1[5] = stride_elems;                                       // stride0 lo
  g1[6] = 0u;                                                 // stride0 hi | stride1 lo
  g1[7] = 0u;                                                 // stride1 hi
  u32x4 gz = {0u, 0u, 0u, 0u};                                // groups 2/3 unused (2-D)
  asm volatile("tensor_load_to_lds %0, %1, %2, %3"
               :: "s"(g0), "s"(g1), "s"(gz), "s"(gz)
               : "memory");
}

// ---------- precision / layout conversion ----------
__global__ void cvt_cast(const float* __restrict__ in, bf16_t* __restrict__ out, int n) {
  int i = blockIdx.x * 256 + threadIdx.x;
  if (i < n) out[i] = f2b(in[i]);
}

// in: K x N row-major (K == 1024), out: N x K row-major (transposed)
__global__ void cvt_transpose(const float* __restrict__ in, bf16_t* __restrict__ out,
                              int N, int total) {
  int i = blockIdx.x * 256 + threadIdx.x;
  if (i >= total) return;
  int k = i & 1023;   // K = 1024
  int n = i >> 10;
  out[i] = f2b(in[(size_t)k * N + n]);
}

// ---------- bf16 WMMA GEMM, TDM-staged + double-buffered LDS ----------
// C(MxN) = A(MxK) * B(KxN), B given transposed (N x K rows).
// MODE 0: scatter qkv -> q[b,h,t,d], k[b,h,t,d], vT[b,h,d,t]   (N = 3072)
// MODE 1: plain f32 output, N = 1024
template <int MODE>
__global__ __launch_bounds__(256) void gemm_wmma(
    const bf16_t* __restrict__ A, const bf16_t* __restrict__ BT,
    float* __restrict__ outF,
    bf16_t* __restrict__ outQ, bf16_t* __restrict__ outK, bf16_t* __restrict__ outV,
    int K) {
  __shared__ __align__(16) bf16_t Alds[2][128 * 32];
  __shared__ __align__(16) bf16_t Blds[2][64 * 32];

  const int tid   = threadIdx.x;
  const int lane  = tid & 31;
  const int wave  = tid >> 5;
  const int waveM = wave & 3;   // 4 waves along M (32 rows each)
  const int waveN = wave >> 2;  // 2 waves along N (32 cols each)
  const int half  = lane >> 4;
  const int l16   = lane & 15;
  const int kb    = half * 8;
  const int mBlock = blockIdx.y * 128;
  const int nBlock = blockIdx.x * 64;

  const bf16_t* Abase = A + (size_t)mBlock * K;
  const bf16_t* Bbase = BT + (size_t)nBlock * K;
  u32 aOff[2], bOff[2];
  aOff[0] = (u32)(uintptr_t)&Alds[0][0];
  aOff[1] = (u32)(uintptr_t)&Alds[1][0];
  bOff[0] = (u32)(uintptr_t)&Blds[0][0];
  bOff[1] = (u32)(uintptr_t)&Blds[1][0];

  v8f acc[2][2] = {};

  // prologue: DMA first tiles into buffer 0
  if (wave == 0) {
    tdm_load_2d(aOff[0], Abase, 128, 32, (u32)K);
    tdm_load_2d(bOff[0], Bbase, 64, 32, (u32)K);
    __builtin_amdgcn_s_wait_tensorcnt(0);
  }
  __syncthreads();

  for (int k0 = 0; k0 < K; k0 += 32) {
    const int cur = (k0 >> 5) & 1;
    const bool more = (k0 + 32) < K;
    if (wave == 0 && more) {  // async DMA of next tiles overlaps this step's WMMA
      tdm_load_2d(aOff[cur ^ 1], Abase + (k0 + 32), 128, 32, (u32)K);
      tdm_load_2d(bOff[cur ^ 1], Bbase + (k0 + 32), 64, 32, (u32)K);
    }

    v16bf af[2], bfr[2];
#pragma unroll
    for (int mt = 0; mt < 2; ++mt)
      af[mt] = load_frag(&Alds[cur][(waveM * 32 + mt * 16 + l16) * 32], kb);
#pragma unroll
    for (int nt = 0; nt < 2; ++nt)
      bfr[nt] = load_frag(&Blds[cur][(waveN * 32 + nt * 16 + l16) * 32], kb);
#pragma unroll
    for (int mt = 0; mt < 2; ++mt)
#pragma unroll
      for (int nt = 0; nt < 2; ++nt)
        acc[mt][nt] = wmma_bf16(af[mt], bfr[nt], acc[mt][nt]);

    if (wave == 0 && more) __builtin_amdgcn_s_wait_tensorcnt(0);
    __syncthreads();
  }

#pragma unroll
  for (int mt = 0; mt < 2; ++mt)
#pragma unroll
    for (int nt = 0; nt < 2; ++nt)
#pragma unroll
      for (int r = 0; r < 8; ++r) {
        int m = mBlock + waveM * 32 + mt * 16 + r + half * 8;
        int n = nBlock + waveN * 32 + nt * 16 + l16;
        float v = acc[mt][nt][r];
        if (MODE == 1) {
          outF[(size_t)m * 1024 + n] = v;
        } else {
          bf16_t bv = f2b(v);
          int part = n >> 10;       // 0:q 1:k 2:v
          int ci   = n & 1023;
          int h    = ci >> 6;
          int d    = ci & 63;
          int b    = m >> 11;
          int t    = m & 2047;
          if (part == 0)
            outQ[(((size_t)(b * 16 + h)) * 2048 + t) * 64 + d] = bv;
          else if (part == 1)
            outK[(((size_t)(b * 16 + h)) * 2048 + t) * 64 + d] = bv;
          else  // v stored transposed (b,h,d,t) for the P*V WMMA B-operand
            outV[(((size_t)(b * 16 + h)) * 64 + d) * 2048 + t] = bv;
        }
      }
}

// ---------- flash attention (one wave = one 16-row q tile) ----------
__global__ __launch_bounds__(128) void attn_wmma(
    const bf16_t* __restrict__ Q, const bf16_t* __restrict__ Km,
    const bf16_t* __restrict__ VT, bf16_t* __restrict__ Y) {
  __shared__ __align__(16) bf16_t plds[4][16][32];

  const int lane = threadIdx.x & 31;
  const int wave = threadIdx.x >> 5;
  const int half = lane >> 4;
  const int l16  = lane & 15;
  const int kb   = half * 8;

  const int blk  = blockIdx.x;
  const int qblk = blk & 31;      // T/64 = 32
  const int bh   = blk >> 5;
  const int h    = bh & 15;
  const int b    = bh >> 4;
  const int qbase = qblk * 64 + wave * 16;

  const size_t headQK = ((size_t)(b * 16 + h)) * 2048 * 64;  // (t,d) layout
  const size_t headV  = ((size_t)(b * 16 + h)) * 64 * 2048;  // (d,t) layout

  const bf16_t* qrow = Q + headQK + (size_t)(qbase + l16) * 64;
  v16bf qf0 = load_frag(qrow, kb);        // d = 0..31
  v16bf qf1 = load_frag(qrow + 32, kb);   // d = 32..63

  v8f o[4] = {};
  float mi[8], li[8];
#pragma unroll
  for (int r = 0; r < 8; ++r) { mi[r] = -3.0e38f; li[r] = 0.0f; }

  const int nkb = (qbase + 47) >> 5;   // key blocks of 32 covering keys <= qbase+15
  for (int j = 0; j < nkb; ++j) {
    const int kt0 = j * 32;
    // prefetch next key block's K and V rows (global_prefetch_b8)
    if (j + 1 < nkb) {
      __builtin_prefetch(Km + headQK + (size_t)(kt0 + 32 + lane) * 64, 0, 3);
      __builtin_prefetch(VT + headV + (size_t)(lane * 2) * 2048 + kt0 + 32, 0, 3);
    }
    // ---- S = (Q K^T) / sqrt(d), two 16-key chunks ----
    v8f s[2];
#pragma unroll
    for (int c = 0; c < 2; ++c) {
      const bf16_t* krow = Km + headQK + (size_t)(kt0 + c * 16 + l16) * 64;
      v8f z = {};
      z = wmma_bf16(qf0, load_frag(krow, kb), z);
      z = wmma_bf16(qf1, load_frag(krow + 32, kb), z);
      s[c] = z;
    }
    // scale + causal mask
#pragma unroll
    for (int c = 0; c < 2; ++c)
#pragma unroll
      for (int r = 0; r < 8; ++r) {
        int tq = qbase + r + half * 8;
        int tk = kt0 + c * 16 + l16;
        float x = s[c][r] * 0.125f;
        s[c][r] = (tk <= tq) ? x : -3.0e38f;
      }
    // ---- online softmax (rows live in 16-lane halves; shfl_xor 1..8 stays inside) ----
    float p0[8], p1[8], scale[8];
#pragma unroll
    for (int r = 0; r < 8; ++r) {
      float cm = fmaxf(s[0][r], s[1][r]);
#pragma unroll
      for (int msk = 1; msk < 16; msk <<= 1) cm = fmaxf(cm, __shfl_xor(cm, msk, 32));
      float mn = fmaxf(mi[r], cm);
      scale[r] = __expf(mi[r] - mn);
      mi[r] = mn;
      p0[r] = __expf(s[0][r] - mn);
      p1[r] = __expf(s[1][r] - mn);
      float rs = p0[r] + p1[r];
#pragma unroll
      for (int msk = 1; msk < 16; msk <<= 1) rs += __shfl_xor(rs, msk, 32);
      li[r] = li[r] * scale[r] + rs;
    }
#pragma unroll
    for (int t = 0; t < 4; ++t)
#pragma unroll
      for (int r = 0; r < 8; ++r) o[t][r] = o[t][r] * scale[r];

    // ---- P: C-layout -> A-layout via wave-private LDS ----
#pragma unroll
    for (int r = 0; r < 8; ++r) {
      plds[wave][r + half * 8][l16]      = f2b(p0[r]);
      plds[wave][r + half * 8][16 + l16] = f2b(p1[r]);
    }
    asm volatile("s_wait_dscnt 0" ::: "memory");
    v16bf pf = load_frag(&plds[wave][l16][0], kb);

    // ---- O += P * V ----
#pragma unroll
    for (int t = 0; t < 4; ++t) {
      const bf16_t* vrow = VT + headV + (size_t)(t * 16 + l16) * 2048 + kt0;
      o[t] = wmma_bf16(pf, load_frag(vrow, kb), o[t]);
    }
    asm volatile("s_wait_dscnt 0" ::: "memory");
  }

#pragma unroll
  for (int t = 0; t < 4; ++t)
#pragma unroll
    for (int r = 0; r < 8; ++r) {
      int tq = qbase + r + half * 8;
      int d  = t * 16 + l16;
      float v = o[t][r] / li[r];
      Y[((size_t)(b * 2048 + tq)) * 1024 + h * 64 + d] = f2b(v);
    }
}

// ---------- launcher ----------
extern "C" void kernel_launch(void* const* d_in, const int* in_sizes, int n_in,
                              void* d_out, int out_size, void* d_ws, size_t ws_size,
                              hipStream_t stream) {
  const float* x     = (const float*)d_in[0];
  const float* wqkv  = (const float*)d_in[1];
  const float* wproj = (const float*)d_in[2];
  float* out = (float*)d_out;

  char* ws = (char*)d_ws;
  size_t off = 0;
  auto take = [&](size_t elems) -> bf16_t* {
    bf16_t* p = (bf16_t*)(ws + off);
    off += (elems * sizeof(bf16_t) + 255) & ~(size_t)255;
    return p;
  };
  const size_t MT = (size_t)4 * 2048;  // 8192 rows
  bf16_t* xb   = take(MT * 1024);                      // x bf16
  bf16_t* wqT  = take((size_t)3072 * 1024);            // w_qkv^T bf16
  bf16_t* wpT  = take((size_t)1024 * 1024);            // w_proj^T bf16
  bf16_t* qb   = take((size_t)4 * 16 * 2048 * 64);     // q (b,h,t,d)
  bf16_t* kbuf = take((size_t)4 * 16 * 2048 * 64);     // k (b,h,t,d)
  bf16_t* vT   = take((size_t)4 * 16 * 64 * 2048);     // v (b,h,d,t)
  bf16_t* yb   = take(MT * 1024);                      // attn output bf16

  {
    int n = (int)(MT * 1024);
    cvt_cast<<<(n + 255) / 256, 256, 0, stream>>>(x, xb, n);
  }
  {
    int total = 3072 * 1024;
    cvt_transpose<<<(total + 255) / 256, 256, 0, stream>>>(wqkv, wqT, 3072, total);
  }
  {
    int total = 1024 * 1024;
    cvt_transpose<<<(total + 255) / 256, 256, 0, stream>>>(wproj, wpT, 1024, total);
  }
  // qkv = x @ w_qkv : M=8192, N=3072, K=1024
  gemm_wmma<0><<<dim3(3072 / 64, 8192 / 128), 256, 0, stream>>>(
      xb, wqT, nullptr, qb, kbuf, vT, 1024);
  // attention: 4*16*(2048/64) = 2048 blocks, 4 waves each
  attn_wmma<<<4 * 16 * (2048 / 64), 128, 0, stream>>>(qb, kbuf, vT, yb);
  // out = y @ w_proj : M=8192, N=1024, K=1024
  gemm_wmma<1><<<dim3(1024 / 64, 8192 / 128), 256, 0, stream>>>(
      yb, wpT, out, nullptr, nullptr, nullptr, 1024);
}
